// StructureEncoder_80616536146707
// MI455X (gfx1250) — compile-verified, matrix operand order checked by reading, not run on hardware
//
#include <hip/hip_runtime.h>
#include <hip/hip_bf16.h>
#include <stdint.h>

typedef __attribute__((ext_vector_type(16))) __bf16 v16bf;
typedef __attribute__((ext_vector_type(8)))  __bf16 v8bf;
typedef __attribute__((ext_vector_type(8)))  float  v8f;
typedef __attribute__((ext_vector_type(4)))  float  v4f;

union U16 { v16bf v; v8bf h[2]; };

#define BATCH 128
#define HDIM  256
#define NNODES 1023

#if __has_builtin(__builtin_amdgcn_sched_group_barrier)
#define SCHED_GROUP(m, n, id) __builtin_amdgcn_sched_group_barrier((m), (n), (id))
#else
#define SCHED_GROUP(m, n, id)
#endif

__device__ __forceinline__ float sigf(float x) {
    return 1.0f / (1.0f + __expf(-x));
}

// ---------------------------------------------------------------------------
// K0: leaf constants + combined gate biases.  biasG layout: [4][256] (f,i,u,o)
// ---------------------------------------------------------------------------
__global__ void init_const_kernel(const float* __restrict__ bf_, const float* __restrict__ b_f,
                                  const float* __restrict__ bi_, const float* __restrict__ b_i,
                                  const float* __restrict__ bu_, const float* __restrict__ b_u,
                                  const float* __restrict__ bo_, const float* __restrict__ b_o,
                                  float* __restrict__ biasG,
                                  float* __restrict__ h_leaf, float* __restrict__ c_leaf) {
    int j = threadIdx.x;
    float Bf = bf_[j] + b_f[j];
    float Bi = bi_[j] + b_i[j];
    float Bu = bu_[j] + b_u[j];
    float Bo = bo_[j] + b_o[j];
    biasG[0 * HDIM + j] = Bf;
    biasG[1 * HDIM + j] = Bi;
    biasG[2 * HDIM + j] = Bu;
    biasG[3 * HDIM + j] = Bo;
    float ig = sigf(Bi);
    float ug = tanhf(Bu);
    float cl = ig * ug;
    c_leaf[j] = cl;
    h_leaf[j] = sigf(Bo) * tanhf(cl);
}

// ---------------------------------------------------------------------------
// Pack W (256 x K, row-major f32) into WMMA B-fragment order (bf16):
//   frag[jt][kc][lane][e]  -> per-lane 32B contiguous, coalesced b128 loads.
// ---------------------------------------------------------------------------
__global__ void pack_frag_kernel(const float* __restrict__ W, __bf16* __restrict__ frag, int K) {
    int idx = blockIdx.x * blockDim.x + threadIdx.x;   // [0, 256*K)
    int jt   = idx / (K * 16);
    int rem  = idx - jt * (K * 16);
    int kc   = rem / 512;
    int rem2 = rem - kc * 512;
    int lane = rem2 >> 4;
    int e    = rem2 & 15;
    int n  = lane & 15;
    int kh = lane >> 4;
    int k_local = (e < 8) ? (kh * 8 + e) : (16 + kh * 8 + (e - 8));
    int j = jt * 16 + n;
    int k = kc * 32 + k_local;
    frag[idx] = (__bf16)W[j * K + k];
}

// ---------------------------------------------------------------------------
// A-prep (8-wide vectorized): A[r][k], r = kk*128 + b, K = 512
// ---------------------------------------------------------------------------
__global__ void a_prep_kernel(const __bf16* __restrict__ hChild,
                              const float* __restrict__ h_leaf,
                              const float* __restrict__ embed,
                              int csGlobal, int leaf,
                              __bf16* __restrict__ A, int M) {
    int idx = blockIdx.x * blockDim.x + threadIdx.x;   // [0, M*64)
    int r  = idx >> 6;
    int kq = (idx & 63) << 3;
    int kk = r >> 7;
    int b  = r & 127;
    v8bf out;
    if (kq < HDIM) {
        if (leaf) {
            v4f h0 = *(const v4f*)(h_leaf + kq);
            v4f h1 = *(const v4f*)(h_leaf + kq + 4);
#pragma unroll
            for (int e = 0; e < 4; ++e) {
                out[e]     = (__bf16)(2.0f * h0[e]);
                out[4 + e] = (__bf16)(2.0f * h1[e]);
            }
        } else {
            size_t c0 = ((size_t)((2 * kk) * BATCH + b)) * HDIM + kq;
            v8bf ha = *(const v8bf*)(hChild + c0);
            v8bf hb = *(const v8bf*)(hChild + c0 + BATCH * HDIM);
#pragma unroll
            for (int e = 0; e < 8; ++e)
                out[e] = (__bf16)((float)ha[e] + (float)hb[e]);
        }
    } else {
        int x = kq - HDIM;
        const float* e0 = embed + ((size_t)b * NNODES + (csGlobal + 2 * kk)) * HDIM + x;
        v4f a0 = *(const v4f*)(e0);
        v4f a1 = *(const v4f*)(e0 + 4);
        v4f b0 = *(const v4f*)(e0 + HDIM);
        v4f b1 = *(const v4f*)(e0 + HDIM + 4);
#pragma unroll
        for (int e = 0; e < 4; ++e) {
            out[e]     = (__bf16)(a0[e] + b0[e]);
            out[4 + e] = (__bf16)(a1[e] + b1[e]);
        }
    }
    *(v8bf*)(A + (size_t)r * 512 + kq) = out;
}

// ---------------------------------------------------------------------------
// Fused 4-gate WMMA GEMM + TreeLSTM cell.
// Block = one jt (16 cols) x 8 mt tiles (8 waves, exact grid, no divergence).
// B fragments (4 gates x K=512 x 16 cols = 64KB) staged in LDS, shared by all
// 8 waves. A (global) and B (LDS) register double-buffered; per-K-step group
// order [2 VMEM (A kc+1)][8 DS (B kc+1)][4 WMMA (kc)] -> 1-deep pipeline, so
// each WMMA batch waits only on loads issued one iteration earlier.
// ~112 live VGPRs for acc+frags: no spills (the 2-mt variant spilled).
// ---------------------------------------------------------------------------
__global__ void gates_gemm_kernel(const __bf16* __restrict__ A,       // M x 512
                                  const __bf16* __restrict__ fragG,   // [4][16jt][16kc][32][16]
                                  const float* __restrict__ biasG,
                                  const float* __restrict__ cChild,
                                  const float* __restrict__ c_leaf,
                                  __bf16* __restrict__ h_out,
                                  float* __restrict__ c_out,
                                  int M, int leaf) {
    __shared__ __bf16 smem[4 * 16 * 512];   // 64 KB: [gate][kc][lane][16]

    const int tid  = threadIdx.x;
    const int jt   = blockIdx.y;
    const int mt   = blockIdx.x * 8 + (tid >> 5);
    const int lane = tid & 31;
    const int n    = lane & 15;
    const int kh   = lane >> 4;

    // cooperative stage of B fragments: 4096 x 16B chunks, coalesced
#pragma unroll
    for (int i = 0; i < 16; ++i) {
        int c   = i * 256 + tid;            // [0, 4096)
        int g   = c >> 10;
        int off = (c & 1023) * 8;
        *(v8bf*)(smem + g * 8192 + off) =
            *(const v8bf*)(fragG + (((g << 4) + jt) << 13) + off);
    }
    __syncthreads();

    const __bf16* arow  = A + ((size_t)(mt * 16 + n)) * 512 + kh * 8;
    const __bf16* lbase = smem + lane * 16;

    v8f acc[4] = {{}, {}, {}, {}};
    U16 aC, aN;
    U16 bC[4], bN[4];

#define LOAD_A(kc, dst)                                     \
    do {                                                    \
        (dst).h[0] = *(const v8bf*)(arow + (kc) * 32);      \
        (dst).h[1] = *(const v8bf*)(arow + (kc) * 32 + 16); \
    } while (0)
#define LOAD_B(kc, g, dst)                                  \
    do {                                                    \
        const __bf16* _p = lbase + (g) * 8192 + (kc) * 512; \
        (dst).h[0] = *(const v8bf*)_p;                      \
        (dst).h[1] = *(const v8bf*)(_p + 8);                \
    } while (0)

    LOAD_A(0, aC);
#pragma unroll
    for (int g = 0; g < 4; ++g) LOAD_B(0, g, bC[g]);

#pragma unroll
    for (int kc = 0; kc < 16; ++kc) {
        if (kc < 15) {
            LOAD_A(kc + 1, aN);
#pragma unroll
            for (int g = 0; g < 4; ++g) LOAD_B(kc + 1, g, bN[g]);
        }
#pragma unroll
        for (int g = 0; g < 4; ++g)
            acc[g] = __builtin_amdgcn_wmma_f32_16x16x32_bf16(false, aC.v, false, bC[g].v,
                                                             (short)0, acc[g], false, false);
        if (kc < 15) {
            aC = aN;
#pragma unroll
            for (int g = 0; g < 4; ++g) bC[g] = bN[g];
        }
        // per-iteration pipeline: [2 VMEM reads][8 DS reads][4 WMMA]
        SCHED_GROUP(0x020, 2, 0);   // VMEM read (A prefetch)
        SCHED_GROUP(0x100, 8, 0);   // DS read   (B prefetch)
        SCHED_GROUP(0x008, 4, 0);   // MFMA/WMMA
    }
#undef LOAD_A
#undef LOAD_B

    // ---- fused TreeLSTM cell epilogue ----
    int j = jt * 16 + n;
    float bF = biasG[0 * HDIM + j];
    float bI = biasG[1 * HDIM + j];
    float bU = biasG[2 * HDIM + j];
    float bO = biasG[3 * HDIM + j];
    float clf = 2.0f * c_leaf[j];

#pragma unroll
    for (int v = 0; v < 8; ++v) {
        int r = mt * 16 + v + 8 * kh;   // C layout: lanes 0-15 -> M=v, lanes 16-31 -> M=v+8
        int kk = r >> 7;
        int b  = r & 127;
        float f = sigf(acc[0][v] + bF);
        float i = sigf(acc[1][v] + bI);
        float u = tanhf(acc[2][v] + bU);
        float o = sigf(acc[3][v] + bO);
        float csum;
        if (leaf) {
            csum = clf;
        } else {
            size_t c0 = ((size_t)((2 * kk) * BATCH + b)) * HDIM + j;
            csum = cChild[c0] + cChild[c0 + BATCH * HDIM];
        }
        float cn = i * u + f * csum;
        float hn = o * tanhf(cn);
        size_t oidx = (size_t)r * HDIM + j;
        c_out[oidx] = cn;
        h_out[oidx] = (__bf16)hn;
    }
}

// ---------------------------------------------------------------------------
// Generic 16x16 WMMA row-tile (K templated, frag-packed B), for the root MLP
// ---------------------------------------------------------------------------
template <int K>
__device__ __forceinline__ v8f tile_gemm(const __bf16* __restrict__ A,
                                         const __bf16* __restrict__ frag,
                                         int mt, int jt, int lane) {
    int n  = lane & 15;
    int kh = lane >> 4;
    const __bf16* arow = A + ((size_t)(mt * 16 + n)) * K + kh * 8;
    const __bf16* bbase = frag + jt * ((K / 32) * 512) + lane * 16;
    v8f acc = {};
#pragma unroll
    for (int kc = 0; kc < K / 32; ++kc) {
        U16 a, b;
        a.h[0] = *(const v8bf*)(arow + kc * 32);
        a.h[1] = *(const v8bf*)(arow + kc * 32 + 16);
        b.h[0] = *(const v8bf*)(bbase + kc * 512);
        b.h[1] = *(const v8bf*)(bbase + kc * 512 + 8);
        acc = __builtin_amdgcn_wmma_f32_16x16x32_bf16(false, a.v, false, b.v, (short)0, acc, false, false);
    }
    return acc;
}

// MLP stage 1: z1 = tanh(hr@W1t + bl1) (f32), z2 = relu(hr@W2t + bl2) (bf16)
__global__ void mlp1_kernel(const __bf16* __restrict__ hr,
                            const __bf16* __restrict__ fragW1, const __bf16* __restrict__ fragW2,
                            const float* __restrict__ bl1, const float* __restrict__ bl2,
                            float* __restrict__ z1, __bf16* __restrict__ z2) {
    int wid  = blockIdx.x * (blockDim.x >> 5) + (threadIdx.x >> 5);
    int lane = threadIdx.x & 31;
    int mt = wid >> 4, jt = wid & 15;
    v8f a1 = tile_gemm<256>(hr, fragW1, mt, jt, lane);
    v8f a2 = tile_gemm<256>(hr, fragW2, mt, jt, lane);
    int n = lane & 15, kh = lane >> 4;
    int j = jt * 16 + n;
    float B1 = bl1[j], B2 = bl2[j];
#pragma unroll
    for (int v = 0; v < 8; ++v) {
        int r = mt * 16 + v + 8 * kh;
        size_t o = (size_t)r * HDIM + j;
        z1[o] = tanhf(a1[v] + B1);
        z2[o] = (__bf16)fmaxf(a2[v] + B2, 0.0f);
    }
}

// MLP stage 2: t = z1 + z2@W3t + bl3  -> bf16
__global__ void mlp2_kernel(const __bf16* __restrict__ z2, const __bf16* __restrict__ fragW3,
                            const float* __restrict__ bl3, const float* __restrict__ z1,
                            __bf16* __restrict__ tbuf) {
    int wid  = blockIdx.x * (blockDim.x >> 5) + (threadIdx.x >> 5);
    int lane = threadIdx.x & 31;
    int mt = wid >> 4, jt = wid & 15;
    v8f a = tile_gemm<256>(z2, fragW3, mt, jt, lane);
    int n = lane & 15, kh = lane >> 4;
    int j = jt * 16 + n;
    float B3 = bl3[j];
#pragma unroll
    for (int v = 0; v < 8; ++v) {
        int r = mt * 16 + v + 8 * kh;
        size_t o = (size_t)r * HDIM + j;
        tbuf[o] = (__bf16)(z1[o] + a[v] + B3);
    }
}

// MLP stage 3: out = relu(t@W4t + bl4) -> f32 d_out
__global__ void mlp3_kernel(const __bf16* __restrict__ tbuf, const __bf16* __restrict__ fragW4,
                            const float* __restrict__ bl4, float* __restrict__ out) {
    int wid  = blockIdx.x * (blockDim.x >> 5) + (threadIdx.x >> 5);
    int lane = threadIdx.x & 31;
    int mt = wid >> 4, jt = wid & 15;
    v8f a = tile_gemm<256>(tbuf, fragW4, mt, jt, lane);
    int n = lane & 15, kh = lane >> 4;
    int j = jt * 16 + n;
    float B4 = bl4[j];
#pragma unroll
    for (int v = 0; v < 8; ++v) {
        int r = mt * 16 + v + 8 * kh;
        out[(size_t)r * HDIM + j] = fmaxf(a[v] + B4, 0.0f);
    }
}

// ---------------------------------------------------------------------------
extern "C" void kernel_launch(void* const* d_in, const int* in_sizes, int n_in,
                              void* d_out, int out_size, void* d_ws, size_t ws_size,
                              hipStream_t stream) {
    const float* embed = (const float*)d_in[0];
    const float* Wf = (const float*)d_in[1];
    const float* bf_ = (const float*)d_in[2];
    const float* b_f = (const float*)d_in[3];
    const float* Wi = (const float*)d_in[4];
    const float* bi_ = (const float*)d_in[5];
    const float* b_i = (const float*)d_in[6];
    const float* Wu = (const float*)d_in[7];
    const float* bu_ = (const float*)d_in[8];
    const float* b_u = (const float*)d_in[9];
    const float* Wo = (const float*)d_in[10];
    const float* bo_ = (const float*)d_in[11];
    const float* b_o = (const float*)d_in[12];
    const float* W1 = (const float*)d_in[13];
    const float* bl1 = (const float*)d_in[14];
    const float* W2 = (const float*)d_in[15];
    const float* bl2 = (const float*)d_in[16];
    const float* W3 = (const float*)d_in[17];
    const float* bl3 = (const float*)d_in[18];
    const float* W4 = (const float*)d_in[19];
    const float* bl4 = (const float*)d_in[20];

    // workspace carve-up (256B aligned)
    uintptr_t p = (uintptr_t)d_ws;
    auto carve = [&](size_t bytes) -> void* {
        void* r = (void*)p;
        p = (p + bytes + 255) & ~(uintptr_t)255;
        return r;
    };
    float*  biasG     = (float*)carve(4 * HDIM * sizeof(float));
    float*  h_leaf    = (float*)carve(HDIM * sizeof(float));
    float*  c_leaf    = (float*)carve(HDIM * sizeof(float));
    __bf16* fragGates = (__bf16*)carve((size_t)4 * HDIM * 512 * sizeof(__bf16));    // 1 MB
    __bf16* fragW1    = (__bf16*)carve((size_t)HDIM * 256 * sizeof(__bf16));
    __bf16* fragW2    = (__bf16*)carve((size_t)HDIM * 256 * sizeof(__bf16));
    __bf16* fragW3    = (__bf16*)carve((size_t)HDIM * 256 * sizeof(__bf16));
    __bf16* fragW4    = (__bf16*)carve((size_t)HDIM * 256 * sizeof(__bf16));
    __bf16* Abuf      = (__bf16*)carve((size_t)32768 * 512 * sizeof(__bf16));       // 33.5 MB
    __bf16* h_int     = (__bf16*)carve((size_t)511 * BATCH * HDIM * sizeof(__bf16));// 33.5 MB
    float*  c_int     = (float*)carve((size_t)511 * BATCH * HDIM * sizeof(float));  // 67 MB
    float*  z1        = (float*)carve((size_t)BATCH * HDIM * sizeof(float));
    __bf16* z2        = (__bf16*)carve((size_t)BATCH * HDIM * sizeof(__bf16));
    __bf16* tbuf      = (__bf16*)carve((size_t)BATCH * HDIM * sizeof(__bf16));
    (void)ws_size; (void)n_in; (void)in_sizes; (void)out_size;

    // constants + weight packing
    init_const_kernel<<<1, 256, 0, stream>>>(bf_, b_f, bi_, b_i, bu_, b_u, bo_, b_o,
                                             biasG, h_leaf, c_leaf);
    const int gateFrag = HDIM * 512;   // elems per gate
    pack_frag_kernel<<<(HDIM * 512) / 256, 256, 0, stream>>>(Wf, fragGates + 0 * gateFrag, 512);
    pack_frag_kernel<<<(HDIM * 512) / 256, 256, 0, stream>>>(Wi, fragGates + 1 * gateFrag, 512);
    pack_frag_kernel<<<(HDIM * 512) / 256, 256, 0, stream>>>(Wu, fragGates + 2 * gateFrag, 512);
    pack_frag_kernel<<<(HDIM * 512) / 256, 256, 0, stream>>>(Wo, fragGates + 3 * gateFrag, 512);
    pack_frag_kernel<<<(HDIM * 256) / 256, 256, 0, stream>>>(W1, fragW1, 256);
    pack_frag_kernel<<<(HDIM * 256) / 256, 256, 0, stream>>>(W2, fragW2, 256);
    pack_frag_kernel<<<(HDIM * 256) / 256, 256, 0, stream>>>(W3, fragW3, 256);
    pack_frag_kernel<<<(HDIM * 256) / 256, 256, 0, stream>>>(W4, fragW4, 256);

    // tree levels, deepest internal level first
    for (int l = 8; l >= 0; --l) {
        int cnt = 1 << l;
        int s   = (1 << l) - 1;         // first node of this level
        int cs  = (1 << (l + 1)) - 1;   // first child node
        int M   = cnt * BATCH;
        int leaf = (l == 8) ? 1 : 0;
        const __bf16* hChild = h_int + (size_t)cs * BATCH * HDIM;
        const float*  cChild = c_int + (size_t)cs * BATCH * HDIM;
        __bf16* h_out = h_int + (size_t)s * BATCH * HDIM;
        float*  c_out = c_int + (size_t)s * BATCH * HDIM;

        a_prep_kernel<<<M / 4, 256, 0, stream>>>(hChild, h_leaf, embed, cs, leaf, Abuf, M);
        // grid: x = groups of 8 row tiles (8 waves x 1 mt), y = 16 column tiles
        dim3 ggrid(M / 128, 16);
        gates_gemm_kernel<<<ggrid, 256, 0, stream>>>(Abuf, fragGates, biasG, cChild, c_leaf,
                                                     h_out, c_out, M, leaf);
    }

    // root MLP: hr = h_int rows 0..127
    const __bf16* hr = h_int;
    mlp1_kernel<<<16, 256, 0, stream>>>(hr, fragW1, fragW2, bl1, bl2, z1, z2);
    mlp2_kernel<<<16, 256, 0, stream>>>(z2, fragW3, bl3, z1, tbuf);
    mlp3_kernel<<<16, 256, 0, stream>>>(tbuf, fragW4, bl4, (float*)d_out);
}
